// SequenceDecoderMoE_71004399337801
// MI455X (gfx1250) — matrix operand order repaired and checked by reading, not compiled, not run on hardware
//
#include <hip/hip_runtime.h>
#include <hip/hip_bf16.h>
#include <math.h>

#define B_ 2
#define L_ 1024
#define S_ 1024
#define V_ 32000
#define D_ 512
#define H_ 8
#define HD_ 64
#define F_ 2048
#define E_ 8
#define NL_ 6
#define NT_ (B_*L_)   // 2048 tokens

typedef __attribute__((ext_vector_type(16))) _Float16 v16h;
typedef __attribute__((ext_vector_type(8)))  _Float16 v8h;
typedef __attribute__((ext_vector_type(4)))  _Float16 v4h;
typedef __attribute__((ext_vector_type(8)))  float    v8f;

__device__ __forceinline__ v8f zero8f(){
  v8f z;
  #pragma unroll
  for (int i=0;i<8;++i) z[i]=0.f;
  return z;
}

__device__ __forceinline__ v8f wmma_f16(v16h a, v16h b, v8f c){
  return __builtin_amdgcn_wmma_f32_16x16x32_f16(false, a, false, b, (short)0, c, false, false);
}

// A-matrix fragment (16x32 f16): lane l holds row (l&15); half j holds
// K = (j/8)*16 + (l>>4)*8 + (j%8)  -> two 16B LDS reads at khi and 16+khi.
__device__ __forceinline__ v16h ld_fragA(const _Float16* p, int stride, int row,
                                         int colbase, int khi){
  const _Float16* q = p + row*stride + colbase + khi;
  v8h lo = *(const v8h*)(q);
  v8h hi = *(const v8h*)(q + 16);
  v16h r;
  #pragma unroll
  for (int j=0;j<8;++j){ r[j]=lo[j]; r[8+j]=hi[j]; }
  return r;
}

// B-matrix fragment (32x16 f16) loaded from B^T layout in LDS (Bt[n][k]):
// lane l holds column n = (l&15); half j holds K = (j/8)*16 + (l>>4)*8 + (j%8)
// -> two 16B LDS reads at (l>>4)*8 and 16+(l>>4)*8.
__device__ __forceinline__ v16h ld_fragB(const _Float16* p, int stride, int row,
                                         int colbase, int khl /* (lane>>4)*16 */){
  const _Float16* q = p + row*stride + colbase + khl/2;
  v8h lo = *(const v8h*)(q);
  v8h hi = *(const v8h*)(q + 16);
  v16h r;
  #pragma unroll
  for (int j=0;j<8;++j){ r[j]=lo[j]; r[8+j]=hi[j]; }
  return r;
}

// ------------------------------------------------------------------
// Generic tiled GEMM: C[M,N] = act(A[M,K] @ W[K,N] + bias)
// Tile 128x128, 8 waves (4 in M x 2 in N), each wave 32x64 = 2x4 frags.
// mode 0: plain rows.  mode 1 (MoE up): A row = perm[segStart+i], C row = segStart+i.
// mode 2 (MoE down): A row = segStart+i, C row = perm[segStart+i].
// For MoE modes, blockIdx.z = expert; W/bias offset by e*K*N / e*N.
// ------------------------------------------------------------------
#define BM 128
#define BN 128
#define BK 32
#define LDSS 40   // padded LDS stride in halves (80B -> conflict-free b128 reads)

__global__ __launch_bounds__(256)
void gemm_kernel(const float* __restrict__ A, int lda,
                 const float* __restrict__ W, int ldw,
                 const float* __restrict__ bias,
                 float* __restrict__ C, int ldc,
                 int M, int N, int K,
                 const int* __restrict__ perm,
                 const int* __restrict__ offs,
                 int mode, int act)
{
  __shared__ __align__(16) _Float16 As[BM][LDSS];
  __shared__ __align__(16) _Float16 Wt[BN][LDSS];

  int tid  = threadIdx.x;
  int lane = tid & 31;
  int wid  = tid >> 5;
  int m0 = blockIdx.x * BM;
  int n0 = blockIdx.y * BN;

  int segStart = 0, segLen = M;
  const float* Wp = W;
  const float* bp = bias;
  if (mode != 0){
    int e = blockIdx.z;
    segStart = offs[e];
    segLen   = offs[e+1] - segStart;
    if (m0 >= segLen) return;
    Wp = W + (size_t)e * K * N;
    bp = bias + (size_t)e * N;
  }

  int wm = wid & 3;         // 0..3 -> 32-row band
  int wn = wid >> 2;        // 0..1 -> 64-col band
  int rm = wm * 32;
  int cn = wn * 64;
  int lr  = lane & 15;
  int khi = (lane >> 4) * 8;
  int khl = (lane >> 4) * 16;

  v8f acc[2][4];
  #pragma unroll
  for (int i=0;i<2;++i)
    #pragma unroll
    for (int j=0;j<4;++j) acc[i][j] = zero8f();

  for (int k0 = 0; k0 < K; k0 += BK){
    __syncthreads();
    // stage A tile 128x32 (f32 -> f16), gather / mask per mode
    #pragma unroll
    for (int it = 0; it < 4; ++it){
      int qid = tid + it*256;          // 0..1023 float4-quads
      int row = qid >> 3;
      int kq  = qid & 7;
      float4 f = make_float4(0.f,0.f,0.f,0.f);
      int i = m0 + row;
      if (mode == 0){
        f = *(const float4*)(A + (size_t)i*lda + k0 + kq*4);
      } else if (i < segLen){
        int grow = (mode == 1) ? perm[segStart + i] : (segStart + i);
        f = *(const float4*)(A + (size_t)grow*lda + k0 + kq*4);
      }
      v4h hv; hv[0]=(_Float16)f.x; hv[1]=(_Float16)f.y; hv[2]=(_Float16)f.z; hv[3]=(_Float16)f.w;
      *(v4h*)(&As[row][kq*4]) = hv;
    }
    // stage W tile 32(k) x 128(n), transposed -> Wt[n][k]
    #pragma unroll
    for (int it = 0; it < 4; ++it){
      int qid = tid + it*256;          // 0..1023 float4-quads
      int kr = qid >> 5;               // 32 quads per k-row
      int nq = qid & 31;
      float4 f = *(const float4*)(Wp + (size_t)(k0+kr)*ldw + n0 + nq*4);
      Wt[nq*4+0][kr] = (_Float16)f.x;
      Wt[nq*4+1][kr] = (_Float16)f.y;
      Wt[nq*4+2][kr] = (_Float16)f.z;
      Wt[nq*4+3][kr] = (_Float16)f.w;
    }
    __syncthreads();

    v16h af0 = ld_fragA(&As[0][0], LDSS, rm + lr,      0, khi);
    v16h af1 = ld_fragA(&As[0][0], LDSS, rm + 16 + lr, 0, khi);
    #pragma unroll
    for (int j=0;j<4;++j){
      v16h bf = ld_fragB(&Wt[0][0], LDSS, cn + j*16 + lr, 0, khl);
      acc[0][j] = wmma_f16(af0, bf, acc[0][j]);
      acc[1][j] = wmma_f16(af1, bf, acc[1][j]);
    }
  }

  // epilogue: C/D layout -> lane row = r + 8*(lane>>4), col = lane&15
  #pragma unroll
  for (int i=0;i<2;++i){
    #pragma unroll
    for (int j=0;j<4;++j){
      #pragma unroll
      for (int r=0;r<8;++r){
        int lrow = rm + i*16 + r + 8*(lane>>4);
        int lcol = cn + j*16 + (lane & 15);
        int gi = m0 + lrow;
        if (mode != 0 && gi >= segLen) continue;
        int grow = (mode == 2) ? perm[segStart + gi]
                 : (mode == 1) ? (segStart + gi) : gi;
        float vv = acc[i][j][r] + bp[n0 + lcol];
        if (act) vv = 0.5f * vv * (1.0f + erff(vv * 0.70710678118654752f)); // exact GELU
        C[(size_t)grow * ldc + n0 + lcol] = vv;
      }
    }
  }
}

// ------------------------------------------------------------------
// Fused flash attention. 4 waves/block, 64 query rows, 32-key chunks.
// Q/K/V are [B, len, D] f32, head slice h*64. O = softmax(QK^T/8 (+causal)) V
// ------------------------------------------------------------------
__global__ __launch_bounds__(128)
void attn_kernel(const float* __restrict__ Qb, const float* __restrict__ Kb,
                 const float* __restrict__ Vb, float* __restrict__ O,
                 int kvLen, int causal)
{
  __shared__ __align__(16) _Float16 Qh[64][72];
  __shared__ __align__(16) _Float16 Kh[32][72];
  __shared__ __align__(16) _Float16 Vt[64][40];
  __shared__ __align__(16) _Float16 Pw[4][16][40];

  int tid  = threadIdx.x;
  int lane = tid & 31;
  int wid  = tid >> 5;
  int lr   = lane & 15;
  int khi  = (lane >> 4) * 8;
  int khl  = (lane >> 4) * 16;
  int qbase = blockIdx.x * 64;
  int bh = blockIdx.y;
  int b = bh / H_;
  int h = bh % H_;

  // stage Q tile [64, 64] f32->f16
  #pragma unroll
  for (int it=0; it<8; ++it){
    int qid = tid + it*128;
    int row = qid >> 4;
    int c4  = qid & 15;
    float4 f = *(const float4*)(Qb + ((size_t)(b*L_ + qbase + row))*D_ + h*HD_ + c4*4);
    v4h hv; hv[0]=(_Float16)f.x; hv[1]=(_Float16)f.y; hv[2]=(_Float16)f.z; hv[3]=(_Float16)f.w;
    *(v4h*)(&Qh[row][c4*4]) = hv;
  }
  __syncthreads();

  v16h aQ0 = ld_fragA(&Qh[0][0], 72, wid*16 + lr,  0, khi);
  v16h aQ1 = ld_fragA(&Qh[0][0], 72, wid*16 + lr, 32, khi);

  float m[8], ssum[8];
  v8f accO[4];
  #pragma unroll
  for (int n=0;n<4;++n) accO[n] = zero8f();
  #pragma unroll
  for (int r=0;r<8;++r){ m[r] = -1e30f; ssum[r] = 0.f; }

  int kend = causal ? (qbase + 64) : kvLen;
  for (int kbeg = 0; kbeg < kend; kbeg += 32){
    __syncthreads();
    // stage K [32,64] row-major (= B^T for QK^T), V transposed -> Vt[hd][key]
    #pragma unroll
    for (int it=0; it<4; ++it){
      int qid = tid + it*128;
      int row = qid >> 4;
      int c4  = qid & 15;
      size_t base = ((size_t)(b*kvLen + kbeg + row))*D_ + h*HD_ + c4*4;
      float4 f = *(const float4*)(Kb + base);
      v4h hv; hv[0]=(_Float16)f.x; hv[1]=(_Float16)f.y; hv[2]=(_Float16)f.z; hv[3]=(_Float16)f.w;
      *(v4h*)(&Kh[row][c4*4]) = hv;
      float4 g = *(const float4*)(Vb + base);
      Vt[c4*4+0][row] = (_Float16)g.x;
      Vt[c4*4+1][row] = (_Float16)g.y;
      Vt[c4*4+2][row] = (_Float16)g.z;
      Vt[c4*4+3][row] = (_Float16)g.w;
    }
    __syncthreads();

    // scores: 16 q-rows x 32 keys  (2 key subtiles x 2 hd chunks = 4 WMMAs)
    v8f sacc0 = zero8f(), sacc1 = zero8f();
    {
      v16h b0 = ld_fragB(&Kh[0][0], 72, lr,      0, khl);
      v16h b1 = ld_fragB(&Kh[0][0], 72, 16 + lr, 0, khl);
      sacc0 = wmma_f16(aQ0, b0, sacc0);
      sacc1 = wmma_f16(aQ0, b1, sacc1);
      b0 = ld_fragB(&Kh[0][0], 72, lr,      32, khl);
      b1 = ld_fragB(&Kh[0][0], 72, 16 + lr, 32, khl);
      sacc0 = wmma_f16(aQ1, b0, sacc0);
      sacc1 = wmma_f16(aQ1, b1, sacc1);
    }

    float sv[2][8];
    #pragma unroll
    for (int r=0;r<8;++r){
      float s0 = sacc0[r] * 0.125f;
      float s1 = sacc1[r] * 0.125f;
      if (causal){
        int qr  = qbase + wid*16 + r + 8*(lane>>4);
        int k0i = kbeg + (lane & 15);
        if (k0i      > qr) s0 = -1e30f;
        if (k0i + 16 > qr) s1 = -1e30f;
      }
      sv[0][r] = s0; sv[1][r] = s1;
    }

    // online softmax per row (rows live per half-wave; 16-lane butterflies)
    #pragma unroll
    for (int r=0;r<8;++r){
      float mx = fmaxf(sv[0][r], sv[1][r]);
      mx = fmaxf(mx, __shfl_xor(mx, 1, 32));
      mx = fmaxf(mx, __shfl_xor(mx, 2, 32));
      mx = fmaxf(mx, __shfl_xor(mx, 4, 32));
      mx = fmaxf(mx, __shfl_xor(mx, 8, 32));
      float mn = fmaxf(m[r], mx);
      float sc = __expf(m[r] - mn);
      m[r] = mn;
      ssum[r] *= sc;
      #pragma unroll
      for (int n=0;n<4;++n) accO[n][r] *= sc;
      float p0 = __expf(sv[0][r] - mn);
      float p1 = __expf(sv[1][r] - mn);
      float ps = p0 + p1;
      ps += __shfl_xor(ps, 1, 32);
      ps += __shfl_xor(ps, 2, 32);
      ps += __shfl_xor(ps, 4, 32);
      ps += __shfl_xor(ps, 8, 32);
      ssum[r] += ps;
      int prow = r + 8*(lane>>4);
      Pw[wid][prow][(lane&15)]      = (_Float16)p0;
      Pw[wid][prow][16 + (lane&15)] = (_Float16)p1;
    }

    // PV: P[16,32] @ V[32,64]  (K=32 == one WMMA K)
    v16h aP = ld_fragA(&Pw[wid][0][0], 40, lr, 0, khi);
    #pragma unroll
    for (int n=0;n<4;++n){
      v16h bV = ld_fragB(&Vt[0][0], 40, n*16 + lr, 0, khl);
      accO[n] = wmma_f16(aP, bV, accO[n]);
    }
  }

  #pragma unroll
  for (int n=0;n<4;++n){
    #pragma unroll
    for (int r=0;r<8;++r){
      int qr = qbase + wid*16 + r + 8*(lane>>4);
      O[((size_t)(b*L_ + qr))*D_ + h*HD_ + n*16 + (lane&15)] = accO[n][r] / ssum[r];
    }
  }
}

// ------------------------------------------------------------------
// Elementwise / routing kernels
// ------------------------------------------------------------------
__global__ void embed_kernel(const int* __restrict__ tok,
                             const float* __restrict__ te,
                             const float* __restrict__ pe,
                             float* __restrict__ x){
  int i = blockIdx.x*256 + threadIdx.x;   // over NT_*D_
  int t = i >> 9;
  int c = i & (D_-1);
  int l = t & (L_-1);
  int tk = tok[t];
  x[i] = te[(size_t)tk*D_ + c] + pe[(size_t)l*D_ + c];
}

__global__ __launch_bounds__(256)
void add_ln_kernel(float* __restrict__ x, const float* __restrict__ y,
                   const float* __restrict__ g, const float* __restrict__ bta){
  __shared__ float red[256];
  int row = blockIdx.x;
  int tid = threadIdx.x;
  size_t base = (size_t)row*D_;
  float v0 = x[base + tid]       + y[base + tid];
  float v1 = x[base + 256 + tid] + y[base + 256 + tid];
  red[tid] = v0 + v1; __syncthreads();
  for (int o=128;o>0;o>>=1){ if (tid<o) red[tid]+=red[tid+o]; __syncthreads(); }
  float mean = red[0] * (1.0f/512.0f);
  __syncthreads();
  float d0 = v0-mean, d1 = v1-mean;
  red[tid] = d0*d0 + d1*d1; __syncthreads();
  for (int o=128;o>0;o>>=1){ if (tid<o) red[tid]+=red[tid+o]; __syncthreads(); }
  float rstd = rsqrtf(red[0] * (1.0f/512.0f) + 1e-5f);
  x[base + tid]       = d0*rstd*g[tid]     + bta[tid];
  x[base + 256 + tid] = d1*rstd*g[256+tid] + bta[256+tid];
}

__global__ void gate_kernel(const float* __restrict__ x, const float* __restrict__ gw,
                            const float* __restrict__ gb, int* __restrict__ idx){
  int t = blockIdx.x*256 + threadIdx.x;
  if (t >= NT_) return;
  float s[E_];
  #pragma unroll
  for (int e=0;e<E_;++e) s[e] = gb[e];
  for (int d=0; d<D_; ++d){
    float xv = x[(size_t)t*D_ + d];
    #pragma unroll
    for (int e=0;e<E_;++e) s[e] += xv * gw[d*E_ + e];
  }
  float best = s[0]; int bi = 0;
  #pragma unroll
  for (int e=1;e<E_;++e) if (s[e] > best){ best = s[e]; bi = e; }
  idx[t] = bi;
}

__global__ void zero_kernel(int* p, int n){
  int i = blockIdx.x*256 + threadIdx.x;
  if (i < n) p[i] = 0;
}
__global__ void count_kernel(const int* __restrict__ idx, int* __restrict__ counts){
  int t = blockIdx.x*256 + threadIdx.x;
  if (t < NT_) atomicAdd(&counts[idx[t]], 1);
}
__global__ void offs_kernel(const int* __restrict__ counts, int* __restrict__ offs,
                            int* __restrict__ cursor){
  if (threadIdx.x == 0){
    int a = 0;
    for (int e=0;e<E_;++e){ offs[e] = a; cursor[e] = a; a += counts[e]; }
    offs[E_] = a;
  }
}
__global__ void scatter_kernel(const int* __restrict__ idx, int* __restrict__ cursor,
                               int* __restrict__ perm){
  int t = blockIdx.x*256 + threadIdx.x;
  if (t < NT_){
    int p = atomicAdd(&cursor[idx[t]], 1);
    perm[p] = t;
  }
}

// ------------------------------------------------------------------
static inline void launch_gemm(const float* A,int lda,const float* W,int ldw,
                               const float* bias,float* C,int ldc,
                               int M,int N,int K,int act,
                               const int* perm,const int* offs,int mode,int gz,
                               hipStream_t s){
  dim3 grid(M/BM, N/BN, gz);
  gemm_kernel<<<grid, dim3(256), 0, s>>>(A,lda,W,ldw,bias,C,ldc,M,N,K,perm,offs,mode,act);
}

extern "C" void kernel_launch(void* const* d_in, const int* in_sizes, int n_in,
                              void* d_out, int out_size, void* d_ws, size_t ws_size,
                              hipStream_t stream) {
  (void)in_sizes; (void)n_in; (void)out_size; (void)ws_size;
  const int*   tok       = (const int*)  d_in[0];
  const float* memory    = (const float*)d_in[1];
  const float* tok_embed = (const float*)d_in[2];
  const float* pos_embed = (const float*)d_in[3];
  const float* sa_wq = (const float*)d_in[4];
  const float* sa_wk = (const float*)d_in[5];
  const float* sa_wv = (const float*)d_in[6];
  const float* sa_wo = (const float*)d_in[7];
  const float* sa_bq = (const float*)d_in[8];
  const float* sa_bk = (const float*)d_in[9];
  const float* sa_bv = (const float*)d_in[10];
  const float* sa_bo = (const float*)d_in[11];
  const float* ca_wq = (const float*)d_in[12];
  const float* ca_wk = (const float*)d_in[13];
  const float* ca_wv = (const float*)d_in[14];
  const float* ca_wo = (const float*)d_in[15];
  const float* ca_bq = (const float*)d_in[16];
  const float* ca_bk = (const float*)d_in[17];
  const float* ca_bv = (const float*)d_in[18];
  const float* ca_bo = (const float*)d_in[19];
  const float* gate_w = (const float*)d_in[20];
  const float* gate_b = (const float*)d_in[21];
  const float* moe_w1 = (const float*)d_in[22];
  const float* moe_b1 = (const float*)d_in[23];
  const float* moe_w2 = (const float*)d_in[24];
  const float* moe_b2 = (const float*)d_in[25];
  const float* ln1_g = (const float*)d_in[26];
  const float* ln1_b = (const float*)d_in[27];
  const float* ln2_g = (const float*)d_in[28];
  const float* ln2_b = (const float*)d_in[29];
  const float* ln3_g = (const float*)d_in[30];
  const float* ln3_b = (const float*)d_in[31];
  const float* out_w = (const float*)d_in[32];
  const float* out_b = (const float*)d_in[33];

  char* ws = (char*)d_ws;
  const size_t XB = (size_t)NT_*D_*sizeof(float);  // 4 MB
  float* x  = (float*)(ws + 0*XB);
  float* t0 = (float*)(ws + 1*XB);
  float* t1 = (float*)(ws + 2*XB);
  float* qb = (float*)(ws + 3*XB);
  float* kb = (float*)(ws + 4*XB);
  float* vb = (float*)(ws + 5*XB);
  float* hb = (float*)(ws + 6*XB);                 // [2048, 2048] = 16 MB
  char*  ip = ws + 6*XB + (size_t)NT_*F_*sizeof(float);
  int* idx    = (int*)(ip);
  int* perm   = (int*)(ip + 8192);
  int* counts = (int*)(ip + 16384);
  int* offs   = (int*)(ip + 16384 + 256);
  int* cursor = (int*)(ip + 16384 + 512);

  // embed + positional
  embed_kernel<<<(NT_*D_)/256, 256, 0, stream>>>(tok, tok_embed, pos_embed, x);

  dim3 agrid(L_/64, B_*H_);
  for (int l = 0; l < NL_; ++l){
    const size_t wo = (size_t)l*D_*D_, bo = (size_t)l*D_;
    // ---- self attention (causal) ----
    launch_gemm(x, D_, sa_wq+wo, D_, sa_bq+bo, qb, D_, NT_, D_, D_, 0, 0, 0, 0, 1, stream);
    launch_gemm(x, D_, sa_wk+wo, D_, sa_bk+bo, kb, D_, NT_, D_, D_, 0, 0, 0, 0, 1, stream);
    launch_gemm(x, D_, sa_wv+wo, D_, sa_bv+bo, vb, D_, NT_, D_, D_, 0, 0, 0, 0, 1, stream);
    attn_kernel<<<agrid, 128, 0, stream>>>(qb, kb, vb, t0, L_, 1);
    launch_gemm(t0, D_, sa_wo+wo, D_, sa_bo+bo, t1, D_, NT_, D_, D_, 0, 0, 0, 0, 1, stream);
    add_ln_kernel<<<NT_, 256, 0, stream>>>(x, t1, ln1_g+bo, ln1_b+bo);
    // ---- cross attention ----
    launch_gemm(x,      D_, ca_wq+wo, D_, ca_bq+bo, qb, D_, NT_, D_, D_, 0, 0, 0, 0, 1, stream);
    launch_gemm(memory, D_, ca_wk+wo, D_, ca_bk+bo, kb, D_, NT_, D_, D_, 0, 0, 0, 0, 1, stream);
    launch_gemm(memory, D_, ca_wv+wo, D_, ca_bv+bo, vb, D_, NT_, D_, D_, 0, 0, 0, 0, 1, stream);
    attn_kernel<<<agrid, 128, 0, stream>>>(qb, kb, vb, t0, S_, 0);
    launch_gemm(t0, D_, ca_wo+wo, D_, ca_bo+bo, t1, D_, NT_, D_, D_, 0, 0, 0, 0, 1, stream);
    add_ln_kernel<<<NT_, 256, 0, stream>>>(x, t1, ln2_g+bo, ln2_b+bo);
    // ---- MoE (top-1 routed, expert-grouped GEMMs) ----
    gate_kernel<<<NT_/256, 256, 0, stream>>>(x, gate_w + (size_t)l*D_*E_, gate_b + (size_t)l*E_, idx);
    zero_kernel<<<1, 256, 0, stream>>>(counts, E_);
    count_kernel<<<NT_/256, 256, 0, stream>>>(idx, counts);
    offs_kernel<<<1, 32, 0, stream>>>(counts, offs, cursor);
    scatter_kernel<<<NT_/256, 256, 0, stream>>>(idx, cursor, perm);
    launch_gemm(x,  D_, moe_w1 + (size_t)l*E_*D_*F_, F_, moe_b1 + (size_t)l*E_*F_,
                hb, F_, NT_, F_, D_, /*gelu*/1, perm, offs, /*mode*/1, E_, stream);
    launch_gemm(hb, F_, moe_w2 + (size_t)l*E_*F_*D_, D_, moe_b2 + (size_t)l*E_*D_,
                t1, D_, NT_, D_, F_, 0, perm, offs, /*mode*/2, E_, stream);
    add_ln_kernel<<<NT_, 256, 0, stream>>>(x, t1, ln3_g+bo, ln3_b+bo);
  }
  // ---- output projection -> logits [B,L,V] ----
  launch_gemm(x, D_, out_w, V_, out_b, (float*)d_out, V_, NT_, V_, D_, 0, 0, 0, 0, 1, stream);
}
